// VRidGGNN_test_54451595378870
// MI455X (gfx1250) — matrix operand order repaired and checked by reading, not compiled, not run on hardware
//
#include <hip/hip_runtime.h>

// ---------------------------------------------------------------------------
// VRidGGNN on MI455X (gfx1250): bf16 WMMA (16x16x32), wave32, dynamic LDS.
// B=4096, T=5, N=8, D=256, E=2, 5 steps.  One WG = 8 batches (M=64 rows).
// ---------------------------------------------------------------------------

typedef __attribute__((ext_vector_type(4)))  float  f32x4;
typedef __attribute__((ext_vector_type(8)))  float  v8f;
typedef __attribute__((ext_vector_type(16))) __bf16 v16bf;

union FragU { v16bf bf; f32x4 f4[2]; };
union PackU { f32x4 f4; unsigned u[4]; };

__device__ __forceinline__ unsigned short f2bf(float x) {
  unsigned u = __float_as_uint(x);
  u += 0x7FFFu + ((u >> 16) & 1u);           // round-to-nearest-even
  return (unsigned short)(u >> 16);
}
__device__ __forceinline__ float bf2f(unsigned short h) {
  return __uint_as_float(((unsigned)h) << 16);
}
__device__ __forceinline__ unsigned pack2(float a, float b) {
  return (unsigned)f2bf(a) | ((unsigned)f2bf(b) << 16);
}
__device__ __forceinline__ float sigmoid_(float x) {
  x = fminf(fmaxf(x, -30.f), 30.f);
  return 1.f / (1.f + __expf(-x));
}
__device__ __forceinline__ float tanh_(float x) {
  x = fminf(fmaxf(x, -15.f), 15.f);
  float e = __expf(2.f * x);
  return (e - 1.f) / (e + 1.f);
}

// A fragment (16x32 bf16): row = lane%16; lanes 0-15 take K 0-15,
// lanes 16-31 take K 16-31 (contiguous, matching B packing).
__device__ __forceinline__ v16bf load_a(const unsigned short* base, int row,
                                        int strideH, int ktLocal, int lane) {
  int kb = ktLocal * 32 + ((lane >> 4) << 4);
  const f32x4* p = (const f32x4*)(base + row * strideH + kb);
  FragU u; u.f4[0] = p[0]; u.f4[1] = p[1];
  return u.bf;
}
// B fragment (32x16 bf16) from pre-packed weights: lane L holds column L%16,
// 16 contiguous K values (K base 0 or 16 by lane half) -> 32 contiguous bytes.
__device__ __forceinline__ v16bf load_b(const unsigned short* __restrict__ pack,
                                        int kt, int nt, int lane) {
  const f32x4* p = (const f32x4*)(pack + ((((kt * 16) + nt) * 32 + lane) << 4));
  FragU u; u.f4[0] = p[0]; u.f4[1] = p[1];
  return u.bf;
}
__device__ __forceinline__ v8f wmma_bf16(v16bf a, v16bf b, v8f c) {
  return __builtin_amdgcn_wmma_f32_16x16x32_bf16(false, a, false, b,
                                                 (short)0, c, false, false);
}

// ---------------------------------------------------------------------------
// Prep kernels: pack fp32 weights into bf16 B-fragment layout (one-time, L2).
// pack[((kt*16+nt)*32 + L)*16 + j] = W[kt*32 + (L/16)*16 + j][nt*16 + L%16]
// ---------------------------------------------------------------------------
__global__ void pack_esplit_kernel(const float* __restrict__ W /*(2,256,256)*/,
                                   unsigned short* __restrict__ dst) {
  int idx = blockIdx.x * blockDim.x + threadIdx.x;   // 131072 elems
  int j  = idx & 15;
  int L  = (idx >> 4) & 31;
  int nt = (idx >> 9) & 15;
  int kt = idx >> 13;                                // 0..15 (e = kt/8)
  int e  = kt >> 3;
  int r  = ((kt & 7) * 32) + ((L >> 4) << 4) + j;
  int c  = nt * 16 + (L & 15);
  dst[idx] = f2bf(W[e * 65536 + r * 256 + c]);
}

__global__ void pack_plain_kernel(const float* __restrict__ W /*(768,256)*/,
                                  unsigned short* __restrict__ dst) {
  int idx = blockIdx.x * blockDim.x + threadIdx.x;   // 196608 elems
  int j  = idx & 15;
  int L  = (idx >> 4) & 31;
  int nt = (idx >> 9) & 15;
  int kt = idx >> 13;                                // 0..23
  int r  = kt * 32 + ((L >> 4) << 4) + j;
  int c  = nt * 16 + (L & 15);
  dst[idx] = f2bf(W[r * 256 + c]);
}

// abias[n][f] = sum_e (sum_{n'} A[n, e*8+n']) * b[e][f]
__global__ void abias_kernel(const float* __restrict__ am,
                             const float* __restrict__ in_b,
                             const float* __restrict__ out_b,
                             float* __restrict__ ab_in,
                             float* __restrict__ ab_out) {
  int idx = blockIdx.x * blockDim.x + threadIdx.x;   // 2048
  int n = idx >> 8, f = idx & 255;
  float s0 = 0.f, s1 = 0.f, s2 = 0.f, s3 = 0.f;
#pragma unroll
  for (int k = 0; k < 8; ++k) {
    s0 += am[n * 32 + k];
    s1 += am[n * 32 + 8 + k];
    s2 += am[n * 32 + 16 + k];
    s3 += am[n * 32 + 24 + k];
  }
  ab_in[idx]  = s0 * in_b[f]  + s1 * in_b[256 + f];
  ab_out[idx] = s2 * out_b[f] + s3 * out_b[256 + f];
}

// ---------------------------------------------------------------------------
// LDS layout (dynamic, 303104 B of the 320KB/WGP):
//   gx_s   f32  [64][260]   66560 B  (persistent state)
//   xm_s   bf16 [64][520]   66560 B  (mixed-x A fragments, K=512)
//   ain_s  bf16 [64][264]   33792 B
//   aout_s bf16 [64][264]   33792 B
//   gxbf_s bf16 [64][264]   33792 B
//   rgx_s  bf16 [64][264]   33792 B
//   zb_s   bf16 [64][264]   33792 B  (xi staging, then z)
//   am_s   f32  [256]        1024 B
// ---------------------------------------------------------------------------
#define SMEM_BYTES 303104

__global__ __launch_bounds__(256)
void ggnn_main_kernel(const float* __restrict__ x,
                      const float* __restrict__ am,
                      const unsigned short* __restrict__ inpack,
                      const unsigned short* __restrict__ outpack,
                      const unsigned short* __restrict__ rgpack,
                      const unsigned short* __restrict__ ugpack,
                      const unsigned short* __restrict__ trpack,
                      const float* __restrict__ abias_in,
                      const float* __restrict__ abias_out,
                      const float* __restrict__ rg_b,
                      const float* __restrict__ ug_b,
                      const float* __restrict__ tr_b,
                      float* __restrict__ out) {
  extern __shared__ char smem[];
  float*          gx_s   = (float*)(smem + 0);
  unsigned short* xm_s   = (unsigned short*)(smem + 66560);
  unsigned short* ain_s  = (unsigned short*)(smem + 133120);
  unsigned short* aout_s = (unsigned short*)(smem + 166912);
  unsigned short* gxbf_s = (unsigned short*)(smem + 200704);
  unsigned short* rgx_s  = (unsigned short*)(smem + 234496);
  unsigned short* zb_s   = (unsigned short*)(smem + 268288);
  float*          am_s   = (float*)(smem + 302080);

  const int t    = threadIdx.x;
  const int lane = t & 31, wave = t >> 5;
  const int ln   = lane & 15, lhalf = lane >> 4;
  const long bbase = (long)blockIdx.x * 8;           // 8 batches per WG

  am_s[t] = am[t];                                   // 8x32 adjacency

  // ---- P0: gx = x[:,0] (f32 master + bf16 A-fragment copy) ----------------
  for (int g = 0; g < 16; ++g) {
    int idx = t + (g << 8);                          // 0..4095
    int m = idx >> 6, c4 = idx & 63;
    int bl = m >> 3, n = m & 7;
    const f32x4* px =
        (const f32x4*)(x + ((((bbase + bl) * 5 + 0) * 8 + n) << 8));
    f32x4 v = px[c4];
    *(f32x4*)(gx_s + m * 260 + (c4 << 2)) = v;
    unsigned* pg = (unsigned*)gxbf_s;
    pg[m * 132 + c4 * 2]     = pack2(v.x, v.y);
    pg[m * 132 + c4 * 2 + 1] = pack2(v.z, v.w);
  }
  __syncthreads();

  for (int s = 0; s < 5; ++s) {
    // ---- stage xi -> zb_s (bf16) ------------------------------------------
    for (int g = 0; g < 16; ++g) {
      int idx = t + (g << 8);
      int m = idx >> 6, c4 = idx & 63;
      int bl = m >> 3, n = m & 7;
      const float* pxf = x + ((((bbase + bl) * 5 + s) * 8 + n) << 8);
      f32x4 v = ((const f32x4*)pxf)[c4];
      if (s < 4) __builtin_prefetch(pxf + 2048 + c4 * 4, 0, 1);  // next step
      unsigned* pz = (unsigned*)zb_s;
      pz[m * 132 + c4 * 2]     = pack2(v.x, v.y);
      pz[m * 132 + c4 * 2 + 1] = pack2(v.z, v.w);
    }
    __syncthreads();

    // ---- a_in / a_out: mix-then-GEMM (K = 512 = two experts) --------------
    for (int phase = 0; phase < 2; ++phase) {
      const int off = phase * 16;                    // A_in vs A_out columns
      // build XM: xm[m][e*256+f] = sum_{n'} am[n,off+e*8+n'] * xi[b,n',f]
      for (int g = 0; g < 16; ++g) {
        int idx = t + (g << 8);                      // 0..4095
        int m = idx >> 6, kc = idx & 63;
        int e = kc >> 5, f8 = (kc & 31) << 3;
        int b8 = m & ~7, n = m & 7;
        float acc[8];
#pragma unroll
        for (int j = 0; j < 8; ++j) acc[j] = 0.f;
#pragma unroll
        for (int np = 0; np < 8; ++np) {
          float w = am_s[n * 32 + off + e * 8 + np];
          f32x4 raw = *(const f32x4*)(zb_s + (b8 + np) * 264 + f8);
          const unsigned* ru = (const unsigned*)&raw;
#pragma unroll
          for (int j = 0; j < 4; ++j) {
            acc[2 * j]     += w * __uint_as_float(ru[j] << 16);
            acc[2 * j + 1] += w * __uint_as_float(ru[j] & 0xFFFF0000u);
          }
        }
        PackU pu;
        pu.u[0] = pack2(acc[0], acc[1]); pu.u[1] = pack2(acc[2], acc[3]);
        pu.u[2] = pack2(acc[4], acc[5]); pu.u[3] = pack2(acc[6], acc[7]);
        *(f32x4*)(xm_s + m * 520 + e * 256 + f8) = pu.f4;
      }
      __syncthreads();

      const unsigned short* pack = phase ? outpack : inpack;
      const float* ab = phase ? abias_out : abias_in;
      unsigned short* dst = phase ? aout_s : ain_s;
      for (int t8 = 0; t8 < 8; ++t8) {
        int tl = t8 * 8 + wave, mt = tl >> 4, nt = tl & 15;
        int col = nt * 16 + ln, arow = mt * 16 + ln;
        v8f c;
#pragma unroll
        for (int i = 0; i < 8; ++i) c[i] = ab[i * 256 + col];
        for (int kt = 0; kt < 16; ++kt)
          c = wmma_bf16(load_a(xm_s, arow, 520, kt, lane),
                        load_b(pack, kt, nt, lane), c);
#pragma unroll
        for (int i = 0; i < 8; ++i)
          dst[(mt * 16 + lhalf * 8 + i) * 264 + col] = f2bf(c[i]);
      }
      __syncthreads();
    }

    // ---- gates r, z: K = 768 over [a_in | a_out | gx] ---------------------
    for (int t8 = 0; t8 < 8; ++t8) {
      int tl = t8 * 8 + wave, mt = tl >> 4, nt = tl & 15;
      int col = nt * 16 + ln, arow = mt * 16 + ln;
      float br = rg_b[col], bz = ug_b[col];
      v8f cr, cz;
#pragma unroll
      for (int i = 0; i < 8; ++i) { cr[i] = br; cz[i] = bz; }
      for (int kt = 0; kt < 24; ++kt) {
        const unsigned short* asrc =
            (kt < 8) ? ain_s : ((kt < 16) ? aout_s : gxbf_s);
        v16bf a = load_a(asrc, arow, 264, kt & 7, lane);
        cr = wmma_bf16(a, load_b(rgpack, kt, nt, lane), cr);
        cz = wmma_bf16(a, load_b(ugpack, kt, nt, lane), cz);
      }
#pragma unroll
      for (int i = 0; i < 8; ++i) {
        int row = mt * 16 + lhalf * 8 + i;
        float r = sigmoid_(cr[i]);
        float z = sigmoid_(cz[i]);
        float g = gx_s[row * 260 + col];
        rgx_s[row * 264 + col] = f2bf(r * g);
        zb_s[row * 264 + col]  = f2bf(z);
      }
    }
    __syncthreads();

    // ---- h_hat + GRU update: K = 768 over [a_in | a_out | r*gx] -----------
    for (int t8 = 0; t8 < 8; ++t8) {
      int tl = t8 * 8 + wave, mt = tl >> 4, nt = tl & 15;
      int col = nt * 16 + ln, arow = mt * 16 + ln;
      float bt = tr_b[col];
      v8f ch;
#pragma unroll
      for (int i = 0; i < 8; ++i) ch[i] = bt;
      for (int kt = 0; kt < 24; ++kt) {
        const unsigned short* asrc =
            (kt < 8) ? ain_s : ((kt < 16) ? aout_s : rgx_s);
        v16bf a = load_a(asrc, arow, 264, kt & 7, lane);
        ch = wmma_bf16(a, load_b(trpack, kt, nt, lane), ch);
      }
#pragma unroll
      for (int i = 0; i < 8; ++i) {
        int row = mt * 16 + lhalf * 8 + i;
        float h  = tanh_(ch[i]);
        float z  = bf2f(zb_s[row * 264 + col]);
        float g  = gx_s[row * 260 + col];
        float gn = fmaf(z, h - g, g);                // (1-z)*g + z*h
        gx_s[row * 260 + col]   = gn;
        gxbf_s[row * 264 + col] = f2bf(gn);
      }
    }
    __syncthreads();
  }

  // ---- epilogue: out[b][f][n] = gx[b][n][f] -------------------------------
  for (int g = 0; g < 64; ++g) {
    int idx = t + (g << 8);                          // 0..16383
    int bl = idx >> 11, rem = idx & 2047;
    int f = rem >> 3, n = rem & 7;
    out[((bbase + bl) << 11) + rem] = gx_s[((bl << 3) + n) * 260 + f];
  }
}

// ---------------------------------------------------------------------------
// Host-side launch
// ---------------------------------------------------------------------------
extern "C" void kernel_launch(void* const* d_in, const int* in_sizes, int n_in,
                              void* d_out, int out_size, void* d_ws,
                              size_t ws_size, hipStream_t stream) {
  const float* x     = (const float*)d_in[0];
  const float* am    = (const float*)d_in[1];
  const float* in_W  = (const float*)d_in[2];
  const float* in_b  = (const float*)d_in[3];
  const float* out_W = (const float*)d_in[4];
  const float* out_b = (const float*)d_in[5];
  const float* rg_W  = (const float*)d_in[6];
  const float* tr_W  = (const float*)d_in[10];
  const float* ug_W  = (const float*)d_in[8];

  char* ws = (char*)d_ws;
  unsigned short* inpack  = (unsigned short*)(ws + 0);        // 262144 B
  unsigned short* outpack = (unsigned short*)(ws + 262144);   // 262144 B
  unsigned short* rgpack  = (unsigned short*)(ws + 524288);   // 393216 B
  unsigned short* ugpack  = (unsigned short*)(ws + 917504);   // 393216 B
  unsigned short* trpack  = (unsigned short*)(ws + 1310720);  // 393216 B
  float* ab_in  = (float*)(ws + 1703936);                     //   8192 B
  float* ab_out = (float*)(ws + 1712128);                     //   8192 B

  // One-time (per call) weight packing fp32 -> bf16 B-fragment layout.
  pack_esplit_kernel<<<512, 256, 0, stream>>>(in_W, inpack);
  pack_esplit_kernel<<<512, 256, 0, stream>>>(out_W, outpack);
  pack_plain_kernel<<<768, 256, 0, stream>>>(rg_W, rgpack);
  pack_plain_kernel<<<768, 256, 0, stream>>>(ug_W, ugpack);
  pack_plain_kernel<<<768, 256, 0, stream>>>(tr_W, trpack);
  abias_kernel<<<8, 256, 0, stream>>>(am, in_b, out_b, ab_in, ab_out);

  // Main fused recurrence: 512 WGs x 256 threads (8 wave32 / WG).
  ggnn_main_kernel<<<512, 256, SMEM_BYTES, stream>>>(
      x, am, inpack, outpack, rgpack, ugpack, trpack, ab_in, ab_out,
      (const float*)d_in[7], (const float*)d_in[9], (const float*)d_in[11],
      (float*)d_out);
}
